// Network_72885595013737
// MI455X (gfx1250) — matrix operand order repaired and checked by reading, not compiled, not run on hardware
//
#include <hip/hip_runtime.h>

// ---------------- problem constants ----------------
#define N_NODES 50000
#define N_EDGES 800000
#define IN_DIM  64
#define H_DIM   128
#define L_LAYERS 3
#define C_CELLS 4
#define K_DIM   32
#define N_OPS   7
#define B_GRAPHS 512
#define OUT_DIM 10
#define CAT_DIM 512   // H*(L+1)
#define D_DIM   1024  // 2*CAT

typedef _Float16 half_t;
typedef __attribute__((ext_vector_type(16))) _Float16 v16h;
typedef __attribute__((ext_vector_type(8)))  _Float16 v8h;
typedef __attribute__((ext_vector_type(8)))  float    v8f;
typedef __attribute__((ext_vector_type(4)))  float    v4f;

// ---------------- WMMA fragment loaders (CDNA5 16x16x32 f16 layouts) ----------------
// A 16x32 (MxK): per lane the 16 elements are two contiguous 8-element runs:
//   lanes 0-15 : row=lane,    K = {k0..k0+7}  and {k0+16..k0+23}
//   lanes 16-31: row=lane-16, K = +8 on both runs
// -> two 16B vector loads per lane (addresses 16B aligned for our strides).
__device__ inline v16h load_a_f32_vec(const float* __restrict__ A, int lda, int row0, int k0) {
  int lane = threadIdx.x & 31;
  int m = lane & 15, hb = (lane >> 4) & 1;
  const float* Ar = A + (size_t)(row0 + m) * lda + k0 + hb * 8;
  v4f a0 = *(const v4f*)Ar;
  v4f a1 = *(const v4f*)(Ar + 4);
  v4f a2 = *(const v4f*)(Ar + 16);
  v4f a3 = *(const v4f*)(Ar + 20);
  v16h a;
#pragma unroll
  for (int i = 0; i < 4; ++i) {
    a[i]      = (half_t)a0[i];
    a[4 + i]  = (half_t)a1[i];
    a[8 + i]  = (half_t)a2[i];
    a[12 + i] = (half_t)a3[i];
  }
  return a;
}
__device__ inline v16h load_a_f16_vec(const half_t* __restrict__ A, int lda, int row0, int k0) {
  int lane = threadIdx.x & 31;
  int m = lane & 15, hb = (lane >> 4) & 1;
  const half_t* Ar = A + (size_t)(row0 + m) * lda + k0 + hb * 8;
  v8h lo = *(const v8h*)Ar;
  v8h hi = *(const v8h*)(Ar + 16);
  v16h a;
#pragma unroll
  for (int i = 0; i < 8; ++i) { a[i] = lo[i]; a[8 + i] = hi[i]; }
  return a;
}
// B 32x16 (KxN) from a K-major transposed f16 weight buffer Bt[col][k] (col stride = kdim):
// per lane: col = col0 + (lane&15); elements = k {k0..k0+7} and {k0+16..k0+23}, +8 for lanes 16-31.
__device__ inline v16h load_bt_f16(const half_t* __restrict__ Bt, int kdim, int k0, int col0) {
  int lane = threadIdx.x & 31;
  int n = lane & 15, hb = (lane >> 4) & 1;
  const half_t* p = Bt + (size_t)(col0 + n) * kdim + k0 + hb * 8;
  v8h lo = *(const v8h*)p;
  v8h hi = *(const v8h*)(p + 16);
  v16h b;
#pragma unroll
  for (int i = 0; i < 8; ++i) { b[i] = lo[i]; b[8 + i] = hi[i]; }
  return b;
}

// ---------------- misc device helpers ----------------
__device__ inline void atomicMaxF(float* addr, float val) {
  if (val >= 0.f) atomicMax((int*)addr, __float_as_int(val));
  else            atomicMin((unsigned int*)addr, __float_as_uint(val));
}

// ---------------- generic kernels ----------------
__global__ void fill_kernel(float* p, float v, long n) {
  long i = (long)blockIdx.x * blockDim.x + threadIdx.x;
  long stride = (long)gridDim.x * blockDim.x;
  for (; i < n; i += stride) p[i] = v;
}
// batched transpose f32[s][R][C] -> f16[s][C][R] (weight prep, runs once per launch)
__global__ void wprep_kernel(const float* __restrict__ Win, half_t* Wt, int R, int C, long total) {
  long i = (long)blockIdx.x * blockDim.x + threadIdx.x;
  if (i >= total) return;
  long rc = (long)R * C;
  long s = i / rc; long rem = i - s * rc;
  int r = (int)(rem / C), c = (int)(rem % C);
  Wt[s * rc + (size_t)c * R + r] = (half_t)Win[i];
}

// Generic f16-compute WMMA GEMM: C = act(A[MxK] @ Bt + bias), Bt = f16 K-major [N][K].
__global__ void __launch_bounds__(256) wmma_gemm_kernel(
    const float* __restrict__ A, int lda, const half_t* __restrict__ Bt,
    const float* __restrict__ bias, float* __restrict__ C, int ldc,
    int M, int N, int K, int act) {
  int mt = blockIdx.x;
  int nt = blockIdx.y * blockDim.y + threadIdx.y;
  if (nt * 16 >= N) return;
  v8f acc = {};
  for (int k0 = 0; k0 < K; k0 += 32) {
    v16h a = load_a_f32_vec(A, lda, mt * 16, k0);
    v16h b = load_bt_f16(Bt, K, k0, nt * 16);
    acc = __builtin_amdgcn_wmma_f32_16x16x32_f16(false, a, false, b, (short)0, acc, false, false);
  }
  int lane = threadIdx.x & 31, n = lane & 15, hb = lane >> 4;
  int col = nt * 16 + n;
  float bs = bias ? bias[col] : 0.f;
#pragma unroll
  for (int r = 0; r < 8; ++r) {
    float v = acc[r] + bs;
    if (act == 1) v = v > 0.f ? v : (expf(v) - 1.f);   // ELU
    else if (act == 2) v = fmaxf(v, 0.f);              // ReLU
    C[(size_t)(mt * 16 + r + hb * 8) * ldc + col] = v;
  }
}

// ---------------- graph / edge kernels ----------------
__global__ void deg_kernel(const int* __restrict__ dst, float* degc, int E) {
  int i = blockIdx.x * blockDim.x + threadIdx.x;
  if (i < E) atomicAdd(&degc[dst[i]], 1.0f);
}
__global__ void deg_finish_kernel(float* degc, float* dinv, int n) {
  int i = blockIdx.x * blockDim.x + threadIdx.x;
  if (i < n) { float d = fmaxf(degc[i], 1.f); degc[i] = d; dinv[i] = rsqrtf(d); }
}
__global__ void vn_init_kernel(float* vn, const float* __restrict__ vn0, long n) {
  long i = (long)blockIdx.x * blockDim.x + threadIdx.x;
  if (i < n) vn[i] = vn0[i % H_DIM];
}
__global__ void add_vn_kernel(float* x, const float* __restrict__ vn, const int* __restrict__ batch, long n) {
  long i = (long)blockIdx.x * blockDim.x + threadIdx.x;
  if (i < n) { long node = i / H_DIM; int h = (int)(i % H_DIM); x[i] += vn[(size_t)batch[node] * H_DIM + h]; }
}
__global__ void catcopy_kernel(const float* __restrict__ x, float* cat, int catoff, long n) {
  long i = (long)blockIdx.x * blockDim.x + threadIdx.x;
  if (i < n) { long node = i / H_DIM; int h = (int)(i % H_DIM); cat[node * CAT_DIM + catoff + h] = x[i]; }
}
// per-node attention projections (one wave / node)
__global__ void __launch_bounds__(256) att_proj_kernel(
    const float* __restrict__ x, const float* __restrict__ a0, const float* __restrict__ a1,
    float* asrc, float* adst, int n) {
  int node = blockIdx.x * blockDim.y + threadIdx.y;
  if (node >= n) return;
  int lane = threadIdx.x;
  const float* xr = x + (size_t)node * H_DIM;
  float s0 = 0.f, s1 = 0.f;
#pragma unroll
  for (int j = 0; j < 4; ++j) { int h = lane + 32 * j; float v = xr[h]; s0 += v * a0[h]; s1 += v * a1[h]; }
  for (int m = 16; m >= 1; m >>= 1) { s0 += __shfl_xor(s0, m, 32); s1 += __shfl_xor(s1, m, 32); }
  if (lane == 0) { asrc[node] = s0; adst[node] = s1; }
}
__global__ void edge_max_kernel(const int* __restrict__ src, const int* __restrict__ dst,
                                const float* __restrict__ asrc, const float* __restrict__ adst,
                                float* m, int E) {
  int i = blockIdx.x * blockDim.x + threadIdx.x;
  if (i >= E) return;
  float e = asrc[src[i]] + adst[dst[i]];
  e = e > 0.f ? e : 0.2f * e;
  atomicMaxF(&m[dst[i]], e);
}
__global__ void m_fix_kernel(float* m, int n) {
  int i = blockIdx.x * blockDim.x + threadIdx.x;
  if (i < n) { float v = m[i]; m[i] = (v >= -3.0e38f && v <= 3.0e38f) ? v : 0.f; }
}
__global__ void edge_exp_kernel(const int* __restrict__ src, const int* __restrict__ dst,
                                const float* __restrict__ asrc, const float* __restrict__ adst,
                                const float* __restrict__ m, float* p, float* ssum, int E) {
  int i = blockIdx.x * blockDim.x + threadIdx.x;
  if (i >= E) return;
  float e = asrc[src[i]] + adst[dst[i]];
  e = e > 0.f ? e : 0.2f * e;
  float pe = expf(e - m[dst[i]]);
  p[i] = pe;
  atomicAdd(&ssum[dst[i]], pe);
}
// fused scatter of all three weighted aggregations; one wave per edge
__global__ void __launch_bounds__(256) edge_agg_kernel(
    const int* __restrict__ src, const int* __restrict__ dst, const float* __restrict__ x,
    const float* __restrict__ dinv, const float* __restrict__ p, const float* __restrict__ ssum,
    float* sum_nb, float* norm_agg, float* attn_agg, int E) {
  int e = blockIdx.x * blockDim.y + threadIdx.y;
  if (e >= E) return;
  int lane = threadIdx.x;
  int s = src[e], d = dst[e];
  float nw = dinv[s] * dinv[d];
  float coef = p[e] / (ssum[d] + 1e-16f);
  const float* xr = x + (size_t)s * H_DIM;
  size_t db = (size_t)d * H_DIM;
#pragma unroll
  for (int j = 0; j < 4; ++j) {
    int h = lane + 32 * j;
    float v = xr[h];
    atomicAdd(&sum_nb[db + h], v);
    atomicAdd(&norm_agg[db + h], nw * v);
    atomicAdd(&attn_agg[db + h], coef * v);
  }
}
// build the 7 aggregation maps as f16 (WMMA A operands)
__global__ void aggs_build_kernel(const float* __restrict__ x, const float* __restrict__ sum_nb,
                                  const float* __restrict__ norm_agg, const float* __restrict__ attn_agg,
                                  const float* __restrict__ degc, half_t* aggs, long n) {
  long i = (long)blockIdx.x * blockDim.x + threadIdx.x;
  if (i >= n) return;
  long node = i / H_DIM;
  float xv = x[i], sn = sum_nb[i], na = norm_agg[i], aa = attn_agg[i];
  float mn = sn / degc[node];
  long NH = n;
  aggs[0 * NH + i] = (half_t)na;
  aggs[1 * NH + i] = (half_t)(xv + sn);
  aggs[2 * NH + i] = (half_t)aa;
  aggs[3 * NH + i] = (half_t)(xv + mn);
  aggs[4 * NH + i] = (half_t)mn;
  aggs[5 * NH + i] = (half_t)na;
  aggs[6 * NH + i] = (half_t)xv;
}

// WMMA einsum 'onh,cohk->ncok' fused with +b_ops, LayerNorm(K), ln affine, softmax(alpha) mix.
// grid (N/16, C), block (32, OPS): wave = one (node-tile, c, o); 8 v_wmma per wave.
__global__ void __launch_bounds__(224) einsum_mix_kernel(
    const half_t* __restrict__ aggs, const half_t* __restrict__ W_opst,
    const float* __restrict__ b_ops, const float* __restrict__ ln_g,
    const float* __restrict__ ln_b, const float* __restrict__ alphas,
    float* __restrict__ x_out, int l) {
  const int mt = blockIdx.x, c = blockIdx.y, o = threadIdx.y;
  const int lane = threadIdx.x, ln = lane & 15, hb = lane >> 4;
  const half_t* Ab = aggs + (size_t)o * N_NODES * H_DIM;
  // W_opst: per (l,c,o) slice stored K-major f16: [k=32][h=128]
  const half_t* Bt = W_opst + (((size_t)l * C_CELLS + c) * N_OPS + o) * (size_t)(H_DIM * K_DIM);
  v8f acc0 = {}, acc1 = {};
  for (int h0 = 0; h0 < H_DIM; h0 += 32) {
    if (h0 + 32 < H_DIM)
      __builtin_prefetch(Ab + (size_t)(mt * 16 + ln) * H_DIM + h0 + 32, 0, 1);
    v16h a  = load_a_f16_vec(Ab, H_DIM, mt * 16, h0);
    v16h b0 = load_bt_f16(Bt, H_DIM, h0, 0);
    v16h b1 = load_bt_f16(Bt, H_DIM, h0, 16);
    acc0 = __builtin_amdgcn_wmma_f32_16x16x32_f16(false, a, false, b0, (short)0, acc0, false, false);
    acc1 = __builtin_amdgcn_wmma_f32_16x16x32_f16(false, a, false, b1, (short)0, acc1, false, false);
  }
  // DARTS mixing weight w[c,o] = softmax(alphas[l,c,:])[o]
  const float* al = alphas + ((size_t)l * C_CELLS + c) * N_OPS;
  float mx = al[0];
#pragma unroll
  for (int i = 1; i < N_OPS; ++i) mx = fmaxf(mx, al[i]);
  float ssw = 0.f, eo = 0.f;
#pragma unroll
  for (int i = 0; i < N_OPS; ++i) { float ee = expf(al[i] - mx); ssw += ee; if (i == o) eo = ee; }
  float w_co = eo / ssw;
  const float* bop = b_ops + (((size_t)l * C_CELLS + c) * N_OPS + o) * K_DIM;
  const float* lg = ln_g + ((size_t)l * C_CELLS + c) * K_DIM;
  const float* lb = ln_b + ((size_t)l * C_CELLS + c) * K_DIM;
  int k0 = ln, k1 = 16 + ln;
  float bop0 = bop[k0], bop1 = bop[k1];
  float lg0 = lg[k0], lg1 = lg[k1], lb0 = lb[k0], lb1 = lb[k1];
#pragma unroll
  for (int r = 0; r < 8; ++r) {
    float v0 = acc0[r] + bop0;
    float v1 = acc1[r] + bop1;
    float s = v0 + v1, q = v0 * v0 + v1 * v1;
    for (int msk = 8; msk >= 1; msk >>= 1) { s += __shfl_xor(s, msk, 32); q += __shfl_xor(q, msk, 32); }
    float mean = s * (1.f / 32.f);
    float var = q * (1.f / 32.f) - mean * mean;
    float rs = rsqrtf(var + 1e-5f);
    v0 = ((v0 - mean) * rs) * lg0 + lb0;
    v1 = ((v1 - mean) * rs) * lg1 + lb1;
    int row = mt * 16 + r + hb * 8;
    float* xr = x_out + (size_t)row * H_DIM + c * K_DIM;
    atomicAdd(xr + k0, w_co * v0);
    atomicAdd(xr + k1, w_co * v1);
  }
}

// ---------------- batch-norm kernels ----------------
// coalesced column accumulation: blocks stream rows, LDS partials, atomic flush
__global__ void __launch_bounds__(256) colaccum_kernel(const float* __restrict__ X, int rows, float* acc) {
  __shared__ float ss[H_DIM], sq[H_DIM];
  int tid = threadIdx.x;
  if (tid < H_DIM) { ss[tid] = 0.f; sq[tid] = 0.f; }
  __syncthreads();
  int col = tid & (H_DIM - 1);
  int sub = tid >> 7;              // 0/1: two rows in flight per block
  float s = 0.f, q = 0.f;
  for (long r = (long)blockIdx.x * 2 + sub; r < rows; r += (long)gridDim.x * 2) {
    float v = X[r * H_DIM + col];
    s += v; q += v * v;
  }
  atomicAdd(&ss[col], s);
  atomicAdd(&sq[col], q);
  __syncthreads();
  if (tid < H_DIM) { atomicAdd(&acc[tid], ss[tid]); atomicAdd(&acc[H_DIM + tid], sq[tid]); }
}
__global__ void colfinish_kernel(const float* __restrict__ acc, int rows, float* stats) {
  int c = blockIdx.x * blockDim.x + threadIdx.x;
  if (c >= H_DIM) return;
  float mean = acc[c] / rows;
  stats[c] = mean;
  stats[H_DIM + c] = acc[H_DIM + c] / rows - mean * mean;
}
__global__ void bn_apply_kernel(float* X, const float* __restrict__ stats,
                                const float* __restrict__ g, const float* __restrict__ b,
                                float* cat, int catoff, long n) {
  long i = (long)blockIdx.x * blockDim.x + threadIdx.x;
  if (i >= n) return;
  int h = (int)(i % H_DIM); long node = i / H_DIM;
  float v = (X[i] - stats[h]) * rsqrtf(stats[H_DIM + h] + 1e-5f) * g[h] + b[h];
  X[i] = v;
  cat[node * CAT_DIM + catoff + h] = v;
}
// BN + ReLU for small (<=512 row) matrices, fused stats+apply (one block per column)
__global__ void colbn_relu_kernel(float* X, int rows, int cols,
                                  const float* __restrict__ g, const float* __restrict__ b) {
  int col = blockIdx.x, tid = threadIdx.x;
  __shared__ float ss[256], sq[256];
  __shared__ float smean, svar;
  float vals[2]; int nv = 0;
  float s = 0.f, q = 0.f;
  for (int r = tid; r < rows && nv < 2; r += 256) { float v = X[(size_t)r * cols + col]; vals[nv++] = v; s += v; q += v * v; }
  ss[tid] = s; sq[tid] = q; __syncthreads();
  for (int st = 128; st >= 1; st >>= 1) {
    if (tid < st) { ss[tid] += ss[tid + st]; sq[tid] += sq[tid + st]; }
    __syncthreads();
  }
  if (tid == 0) { smean = ss[0] / rows; svar = sq[0] / rows - smean * smean; }
  __syncthreads();
  float rs = rsqrtf(svar + 1e-5f);
  int k = 0;
  for (int r = tid; r < rows && k < 2; r += 256) {
    float v = (vals[k++] - smean) * rs * g[col] + b[col];
    X[(size_t)r * cols + col] = fmaxf(v, 0.f);
  }
}

// ---------------- virtual node + pooling + head ----------------
__global__ void vt_scatter_kernel(const float* __restrict__ x, const int* __restrict__ batch, float* vt, long n) {
  long i = (long)blockIdx.x * blockDim.x + threadIdx.x;
  if (i >= n) return;
  long node = i / H_DIM; int h = (int)(i % H_DIM);
  atomicAdd(&vt[(size_t)batch[node] * H_DIM + h], x[i]);
}
__global__ void vt_addvn_kernel(float* vt, const float* __restrict__ vn, long n) {
  long i = (long)blockIdx.x * blockDim.x + threadIdx.x;
  if (i < n) vt[i] += vn[i];
}
__global__ void cnt_kernel(const int* __restrict__ batch, float* cnt, int n) {
  int i = blockIdx.x * blockDim.x + threadIdx.x;
  if (i < n) atomicAdd(&cnt[batch[i]], 1.f);
}
__global__ void pool_scatter_kernel(const float* __restrict__ cat, const int* __restrict__ batch,
                                    float* gsum, float* gmax, long n) {
  long i = (long)blockIdx.x * blockDim.x + threadIdx.x;
  if (i >= n) return;
  long node = i / CAT_DIM; int j = (int)(i % CAT_DIM);
  float v = cat[i];
  size_t b = (size_t)batch[node] * CAT_DIM + j;
  atomicAdd(&gsum[b], v);
  atomicMaxF(&gmax[b], v);
}
__global__ void pool_finish_kernel(const float* __restrict__ gsum, const float* __restrict__ gmax,
                                   const float* __restrict__ cnt, float* pooled, long n) {
  long i = (long)blockIdx.x * blockDim.x + threadIdx.x;
  if (i >= n) return;
  int b = (int)(i / CAT_DIM), j = (int)(i % CAT_DIM);
  float c = fmaxf(cnt[b], 1.f);
  pooled[(size_t)b * D_DIM + j] = gsum[i] / c;
  float gm = gmax[i];
  pooled[(size_t)b * D_DIM + CAT_DIM + j] = (gm >= -3.0e38f && gm <= 3.0e38f) ? gm : 0.f;
}
__global__ void __launch_bounds__(256) head_kernel(const float* __restrict__ pooled,
                                                   const float* __restrict__ W, const float* __restrict__ mask,
                                                   const float* __restrict__ hb, float* out) {
  int wid = blockIdx.x * blockDim.y + threadIdx.y;
  if (wid >= B_GRAPHS * OUT_DIM) return;
  int b = wid / OUT_DIM, j = wid % OUT_DIM;
  int lane = threadIdx.x;
  float s = 0.f;
  for (int d = lane; d < D_DIM; d += 32)
    s += pooled[(size_t)b * D_DIM + d] * W[(size_t)d * OUT_DIM + j] * mask[(size_t)d * OUT_DIM + j];
  for (int m = 16; m >= 1; m >>= 1) s += __shfl_xor(s, m, 32);
  if (lane == 0) out[(size_t)b * OUT_DIM + j] = s + hb[j];
}

// ---------------- host orchestration ----------------
extern "C" void kernel_launch(void* const* d_in, const int* in_sizes, int n_in,
                              void* d_out, int out_size, void* d_ws, size_t ws_size,
                              hipStream_t stream) {
  (void)in_sizes; (void)n_in; (void)out_size; (void)ws_size;
  const float* x_in   = (const float*)d_in[0];
  const int*   ei     = (const int*)d_in[1];
  const int*   batch  = (const int*)d_in[2];
  const float* lin1_W = (const float*)d_in[3];
  const float* lin1_b = (const float*)d_in[4];
  const float* vn0    = (const float*)d_in[5];
  const float* W_ops  = (const float*)d_in[6];
  const float* b_ops  = (const float*)d_in[7];
  const float* att_a  = (const float*)d_in[8];
  const float* ln_g   = (const float*)d_in[9];
  const float* ln_b   = (const float*)d_in[10];
  const float* bn_g   = (const float*)d_in[11];
  const float* bn_b   = (const float*)d_in[12];
  const float* vn_W1  = (const float*)d_in[13];
  const float* vn_b1  = (const float*)d_in[14];
  const float* vn_bn1_g = (const float*)d_in[15];
  const float* vn_bn1_b = (const float*)d_in[16];
  const float* vn_W2  = (const float*)d_in[17];
  const float* vn_b2  = (const float*)d_in[18];
  const float* vn_bn2_g = (const float*)d_in[19];
  const float* vn_bn2_b = (const float*)d_in[20];
  const float* alphas = (const float*)d_in[21];
  const float* head_W = (const float*)d_in[22];
  const float* head_b = (const float*)d_in[23];
  const float* head_mask = (const float*)d_in[24];
  float* out = (float*)d_out;

  const int* src = ei;
  const int* dst = ei + N_EDGES;

  const long NH = (long)N_NODES * H_DIM;

  // workspace carve-out (floats; every region is a multiple of 4 floats -> 16B aligned)
  float* W = (float*)d_ws;
  size_t off = 0;
  auto alloc = [&](size_t n) { float* p = W + off; off += n; return p; };
  float* x_cur    = alloc(NH);
  float* cat      = alloc((size_t)N_NODES * CAT_DIM);
  float* degc     = alloc(N_NODES);
  float* dinv     = alloc(N_NODES);
  float* asrc     = alloc(N_NODES);
  float* adst     = alloc(N_NODES);
  float* mbuf     = alloc(N_NODES);
  float* ssum     = alloc(N_NODES);
  float* pbuf     = alloc(N_EDGES);
  float* sum_nb   = alloc(NH);
  float* norm_agg = alloc(NH);
  float* attn_agg = alloc(NH);
  half_t* aggs    = (half_t*)alloc((size_t)N_OPS * NH / 2 + 16);
  float* bnacc    = alloc(2 * H_DIM);
  float* bnstats  = alloc(2 * H_DIM);
  float* vt       = alloc((size_t)B_GRAPHS * H_DIM);
  float* h1       = alloc((size_t)B_GRAPHS * 2 * H_DIM);
  float* vn_buf   = alloc((size_t)B_GRAPHS * H_DIM);
  float* cnt      = alloc(B_GRAPHS);
  float* gsum     = alloc((size_t)B_GRAPHS * CAT_DIM);
  float* gmax     = alloc((size_t)B_GRAPHS * CAT_DIM);
  float* pooled   = alloc((size_t)B_GRAPHS * D_DIM);
  // f16 transposed weight buffers (prep once per launch)
  half_t* lin1t   = (half_t*)alloc((size_t)IN_DIM * H_DIM / 2);                       // [128][64]
  half_t* W_opst  = (half_t*)alloc((size_t)L_LAYERS * C_CELLS * N_OPS * H_DIM * K_DIM / 2); // [l,c,o][32][128]
  half_t* vn_W1t  = (half_t*)alloc((size_t)(L_LAYERS - 1) * H_DIM * 2 * H_DIM / 2);   // [l][256][128]
  half_t* vn_W2t  = (half_t*)alloc((size_t)(L_LAYERS - 1) * 2 * H_DIM * H_DIM / 2);   // [l][128][256]

  auto fillN = [&](float* p, float v, long n) {
    int blocks = (int)((n + 255) / 256); if (blocks > 16384) blocks = 16384;
    fill_kernel<<<blocks, 256, 0, stream>>>(p, v, n);
  };
  const int NT = N_NODES / 16;             // 3125 node tiles
  const int GNH = (int)((NH + 255) / 256); // 25000

  // 0) weight prep: transpose all GEMM weights into f16 K-major B buffers
  {
    long n;
    n = (long)IN_DIM * H_DIM;
    wprep_kernel<<<(int)((n + 255) / 256), 256, 0, stream>>>(lin1_W, lin1t, IN_DIM, H_DIM, n);
    n = (long)L_LAYERS * C_CELLS * N_OPS * H_DIM * K_DIM;
    wprep_kernel<<<(int)((n + 255) / 256), 256, 0, stream>>>(W_ops, W_opst, H_DIM, K_DIM, n);
    n = (long)(L_LAYERS - 1) * H_DIM * 2 * H_DIM;
    wprep_kernel<<<(int)((n + 255) / 256), 256, 0, stream>>>(vn_W1, vn_W1t, H_DIM, 2 * H_DIM, n);
    wprep_kernel<<<(int)((n + 255) / 256), 256, 0, stream>>>(vn_W2, vn_W2t, 2 * H_DIM, H_DIM, n);
  }

  // 1) encoder: x = elu(x @ lin1_W + b)   [WMMA]
  wmma_gemm_kernel<<<dim3(NT, 1), dim3(32, 8), 0, stream>>>(
      x_in, IN_DIM, lin1t, lin1_b, x_cur, H_DIM, N_NODES, H_DIM, IN_DIM, 1);
  catcopy_kernel<<<GNH, 256, 0, stream>>>(x_cur, cat, 0, NH);

  // 2) degrees / gcn-norm
  fillN(degc, 0.f, N_NODES);
  deg_kernel<<<(N_EDGES + 255) / 256, 256, 0, stream>>>(dst, degc, N_EDGES);
  deg_finish_kernel<<<(N_NODES + 255) / 256, 256, 0, stream>>>(degc, dinv, N_NODES);

  // 3) virtual-node init
  vn_init_kernel<<<((long)B_GRAPHS * H_DIM + 255) / 256, 256, 0, stream>>>(vn_buf, vn0, (long)B_GRAPHS * H_DIM);

  for (int l = 0; l < L_LAYERS; ++l) {
    // x += vn[batch]
    add_vn_kernel<<<GNH, 256, 0, stream>>>(x_cur, vn_buf, batch, NH);
    // attention projections + segment softmax
    att_proj_kernel<<<(N_NODES + 7) / 8, dim3(32, 8), 0, stream>>>(
        x_cur, att_a + (size_t)l * 2 * H_DIM, att_a + (size_t)l * 2 * H_DIM + H_DIM, asrc, adst, N_NODES);
    fillN(mbuf, -INFINITY, N_NODES);
    fillN(ssum, 0.f, N_NODES);
    edge_max_kernel<<<(N_EDGES + 255) / 256, 256, 0, stream>>>(src, dst, asrc, adst, mbuf, N_EDGES);
    m_fix_kernel<<<(N_NODES + 255) / 256, 256, 0, stream>>>(mbuf, N_NODES);
    edge_exp_kernel<<<(N_EDGES + 255) / 256, 256, 0, stream>>>(src, dst, asrc, adst, mbuf, pbuf, ssum, N_EDGES);
    // fused edge aggregations (sum / gcn-norm / attention)
    fillN(sum_nb, 0.f, NH); fillN(norm_agg, 0.f, NH); fillN(attn_agg, 0.f, NH);
    edge_agg_kernel<<<(N_EDGES + 7) / 8, dim3(32, 8), 0, stream>>>(
        src, dst, x_cur, dinv, pbuf, ssum, sum_nb, norm_agg, attn_agg, N_EDGES);
    // build 7 aggregation maps (f16 for WMMA)
    aggs_build_kernel<<<GNH, 256, 0, stream>>>(x_cur, sum_nb, norm_agg, attn_agg, degc, aggs, NH);
    // WMMA einsum + bias + LayerNorm + alpha-softmax mix -> x_cur
    fillN(x_cur, 0.f, NH);
    einsum_mix_kernel<<<dim3(NT, C_CELLS), dim3(32, N_OPS), 0, stream>>>(
        aggs, W_opst, b_ops, ln_g, ln_b, alphas, x_cur, l);
    // BatchNorm over nodes (coalesced stats); append to concat buffer
    fillN(bnacc, 0.f, 2 * H_DIM);
    colaccum_kernel<<<256, 256, 0, stream>>>(x_cur, N_NODES, bnacc);
    colfinish_kernel<<<1, 256, 0, stream>>>(bnacc, N_NODES, bnstats);
    bn_apply_kernel<<<GNH, 256, 0, stream>>>(x_cur, bnstats, bn_g + (size_t)l * H_DIM,
                                             bn_b + (size_t)l * H_DIM, cat, (l + 1) * H_DIM, NH);
    if (l < L_LAYERS - 1) {
      // virtual-node MLP (WMMA GEMMs + fused BN/ReLU)
      fillN(vt, 0.f, (long)B_GRAPHS * H_DIM);
      vt_scatter_kernel<<<GNH, 256, 0, stream>>>(x_cur, batch, vt, NH);
      vt_addvn_kernel<<<((long)B_GRAPHS * H_DIM + 255) / 256, 256, 0, stream>>>(vt, vn_buf, (long)B_GRAPHS * H_DIM);
      wmma_gemm_kernel<<<dim3(B_GRAPHS / 16, 2), dim3(32, 8), 0, stream>>>(
          vt, H_DIM, vn_W1t + (size_t)l * 2 * H_DIM * H_DIM,
          vn_b1 + (size_t)l * 2 * H_DIM, h1, 2 * H_DIM, B_GRAPHS, 2 * H_DIM, H_DIM, 0);
      colbn_relu_kernel<<<2 * H_DIM, 256, 0, stream>>>(h1, B_GRAPHS, 2 * H_DIM,
          vn_bn1_g + (size_t)l * 2 * H_DIM, vn_bn1_b + (size_t)l * 2 * H_DIM);
      wmma_gemm_kernel<<<dim3(B_GRAPHS / 16, 1), dim3(32, 8), 0, stream>>>(
          h1, 2 * H_DIM, vn_W2t + (size_t)l * 2 * H_DIM * H_DIM,
          vn_b2 + (size_t)l * H_DIM, vn_buf, H_DIM, B_GRAPHS, H_DIM, 2 * H_DIM, 0);
      colbn_relu_kernel<<<H_DIM, 256, 0, stream>>>(vn_buf, B_GRAPHS, H_DIM,
          vn_bn2_g + (size_t)l * H_DIM, vn_bn2_b + (size_t)l * H_DIM);
    }
  }

  // 4) global mean/max pooling over graphs
  fillN(cnt, 0.f, B_GRAPHS);
  fillN(gsum, 0.f, (long)B_GRAPHS * CAT_DIM);
  fillN(gmax, -INFINITY, (long)B_GRAPHS * CAT_DIM);
  cnt_kernel<<<(N_NODES + 255) / 256, 256, 0, stream>>>(batch, cnt, N_NODES);
  {
    long n = (long)N_NODES * CAT_DIM;
    pool_scatter_kernel<<<(int)((n + 255) / 256), 256, 0, stream>>>(cat, batch, gsum, gmax, n);
  }
  pool_finish_kernel<<<((long)B_GRAPHS * CAT_DIM + 255) / 256, 256, 0, stream>>>(
      gsum, gmax, cnt, pooled, (long)B_GRAPHS * CAT_DIM);

  // 5) masked head
  head_kernel<<<(B_GRAPHS * OUT_DIM + 7) / 8, dim3(32, 8), 0, stream>>>(
      pooled, head_W, head_mask, head_b, out);
}
